// _WaveNet_87608742904259
// MI455X (gfx1250) — matrix operand order repaired and checked by reading, not compiled, not run on hardware
//
#include <hip/hip_runtime.h>

#define B_    4
#define T_    16384
#define BT    65536
#define FIN   64
#define RC    128
#define SC    32
#define FC    256
#define NLAYER 20
#define WAVES 4
#define MROWS 32          // rows per wave (two 16-row WMMA sub-tiles)

typedef __attribute__((ext_vector_type(16))) __bf16 v16bf;
typedef __attribute__((ext_vector_type(8)))  __bf16 v8bf;
typedef __attribute__((ext_vector_type(8)))  float  v8f;

__device__ __forceinline__ __bf16 f2bf(float f) {
  unsigned u = __builtin_bit_cast(unsigned, f);
  u += 0x7fffu + ((u >> 16) & 1u);                 // round-to-nearest-even
  unsigned short h = (unsigned short)(u >> 16);
  return __builtin_bit_cast(__bf16, h);
}

__device__ __forceinline__ v8f wmma_bf(v16bf a, v16bf b, v8f c) {
  // D = A(16x32 bf16) * B(32x16 bf16) + C(16x16 f32)
  return __builtin_amdgcn_wmma_f32_16x16x32_bf16(false, a, false, b, (short)0, c,
                                                 false, false);
}

// A-fragment (16x32, row-major source, ld elements per row).
// lane&15 = M row; per-lane K chunk: [hi*16, hi*16+16).
// Rows with t = t_first + (lane&15) < 0 are the causal zero-pad.
__device__ __forceinline__ v16bf a_frag(const __bf16* base, int t_first, int kbase,
                                        int ld, int lane) {
  const int m = lane & 15, hi = lane >> 4;
  const int t = t_first + m;
  v16bf a;
  if (t >= 0) {
    const __bf16* p = base + (size_t)t * ld + kbase + hi * 16;
    v8bf lo = *(const v8bf*)p;
    v8bf hh = *(const v8bf*)(p + 8);
#pragma unroll
    for (int i = 0; i < 8; ++i) { a[i] = lo[i]; a[i + 8] = hh[i]; }
  } else {
    const __bf16 z = f2bf(0.f);
#pragma unroll
    for (int i = 0; i < 16; ++i) a[i] = z;
  }
  return a;
}

// B-fragment from weights stored transposed as (N,K) row-major (ld = K).
// lane&15 = N column; lo/hi lane halves take K=[kbase,kbase+16)/[kbase+16,kbase+32).
__device__ __forceinline__ v16bf b_frag(const __bf16* wT, int ntile, int kbase,
                                        int ld, int lane) {
  const int n = lane & 15, hi = lane >> 4;
  const __bf16* p = wT + (size_t)(ntile * 16 + n) * ld + kbase + hi * 16;
  return *(const v16bf*)p;
}

// ---------------- prep kernels ----------------
__global__ void kCvt(const float* __restrict__ src, __bf16* __restrict__ dst, int n) {
  for (int i = blockIdx.x * blockDim.x + threadIdx.x; i < n;
       i += gridDim.x * blockDim.x)
    dst[i] = f2bf(src[i]);
}

// (count,K,N) f32 -> (count,N,K) bf16 ; count = blockIdx.y
__global__ void kTr(const float* __restrict__ src, __bf16* __restrict__ dst,
                    int K, int N) {
  const size_t base = (size_t)blockIdx.y * K * N;
  for (int i = blockIdx.x * blockDim.x + threadIdx.x; i < K * N;
       i += gridDim.x * blockDim.x) {
    const int k = i / N, n = i - k * N;
    dst[base + (size_t)n * K + k] = f2bf(src[base + i]);
  }
}

// ---------------- initial causal conv: K=32 taps, 64 -> 128 ----------------
__global__ __launch_bounds__(128) void kInit(const __bf16* __restrict__ Xbf,
    const __bf16* __restrict__ initT, const float* __restrict__ init_b,
    float* __restrict__ res_f, __bf16* __restrict__ res_a,
    float* __restrict__ out_ac) {
  const int lane = threadIdx.x & 31, wave = threadIdx.x >> 5;
  const int m = lane & 15, hi = lane >> 4;
  const int row0 = (blockIdx.x * WAVES + wave) * MROWS;
  const int b = row0 >> 14, t0 = row0 & (T_ - 1);
  const __bf16* Xb = Xbf + (size_t)b * T_ * FIN;

  v8f acc[2][8];
#pragma unroll
  for (int s = 0; s < 2; ++s)
#pragma unroll
    for (int nt = 0; nt < 8; ++nt)
#pragma unroll
      for (int r = 0; r < 8; ++r) acc[s][nt][r] = 0.f;

  for (int k = 0; k < 32; ++k) {
    const int tf = t0 - 31 + k;
    const __bf16* wk = initT + (size_t)k * RC * FIN;
#pragma unroll
    for (int kb = 0; kb < 2; ++kb) {
      v16bf a0 = a_frag(Xb, tf,      kb * 32, FIN, lane);
      v16bf a1 = a_frag(Xb, tf + 16, kb * 32, FIN, lane);
      // software-pipelined B stream: next fragment loads into a fresh register
      // while the current one feeds two WMMAs.
      v16bf bw = b_frag(wk, 0, kb * 32, FIN, lane);
#pragma unroll
      for (int nt = 0; nt < 8; ++nt) {
        v16bf bn;
        if (nt < 7) bn = b_frag(wk, nt + 1, kb * 32, FIN, lane);
        acc[0][nt] = wmma_bf(a0, bw, acc[0][nt]);
        acc[1][nt] = wmma_bf(a1, bw, acc[1][nt]);
        if (nt < 7) bw = bn;
      }
    }
  }
#pragma unroll
  for (int s = 0; s < 2; ++s)
#pragma unroll
    for (int nt = 0; nt < 8; ++nt) {
      const float bias = init_b[nt * 16 + m];
#pragma unroll
      for (int r = 0; r < 8; ++r) {
        const float v = acc[s][nt][r] + bias;
        const size_t idx =
            (size_t)(row0 + s * 16 + r + hi * 8) * RC + nt * 16 + m;
        res_f[idx] = v;
        res_a[idx] = f2bf(v);
      }
    }
  for (int i = lane; i < MROWS * RC; i += 32)        // zero skip accumulator
    out_ac[(size_t)row0 * RC + i] = 0.f;
}

// ---------------- one dilated gated layer ----------------
__global__ __launch_bounds__(128) void kLayer(const __bf16* __restrict__ rin,
    float* __restrict__ res_f, __bf16* __restrict__ rout,
    float* __restrict__ out_ac,
    const __bf16* __restrict__ fT0, const __bf16* __restrict__ fT1,
    const __bf16* __restrict__ gT0, const __bf16* __restrict__ gT1,
    const __bf16* __restrict__ skT, const __bf16* __restrict__ reT,
    const float* __restrict__ fB, const float* __restrict__ gB,
    const float* __restrict__ sB, const float* __restrict__ rB, int dil) {
  __shared__ __bf16 lds[WAVES * MROWS * RC];
  const int lane = threadIdx.x & 31, wave = threadIdx.x >> 5;
  const int m = lane & 15, hi = lane >> 4;
  __bf16* pl = lds + wave * MROWS * RC;
  const int row0 = (blockIdx.x * WAVES + wave) * MROWS;
  const int b = row0 >> 14, t0 = row0 & (T_ - 1);
  const __bf16* rb = rin + (size_t)b * T_ * RC;

  __builtin_prefetch(fT0, 0, 2);                      // global_prefetch_b8
  __builtin_prefetch(gT0, 0, 2);
  __builtin_prefetch(skT, 0, 2);

  v16bf Ac[2][4], Ap[2][4];
#pragma unroll
  for (int s = 0; s < 2; ++s)
#pragma unroll
    for (int kb = 0; kb < 4; ++kb) {
      Ac[s][kb] = a_frag(rb, t0 + s * 16,       kb * 32, RC, lane);   // tap t
      Ap[s][kb] = a_frag(rb, t0 + s * 16 - dil, kb * 32, RC, lane);   // tap t-d
    }

  for (int nt = 0; nt < 8; ++nt) {                    // f,g over all 128 out ch
    const float fb = fB[nt * 16 + m], gb = gB[nt * 16 + m];
    v8f fa[2], ga[2];
#pragma unroll
    for (int s = 0; s < 2; ++s)
#pragma unroll
      for (int r = 0; r < 8; ++r) { fa[s][r] = fb; ga[s][r] = gb; }
#pragma unroll
    for (int kb = 0; kb < 4; ++kb) {
      // one 4x-b128-pair clause per kb group, amortized over 8 WMMAs
      v16bf b0 = b_frag(fT0, nt, kb * 32, RC, lane);
      v16bf b1 = b_frag(fT1, nt, kb * 32, RC, lane);
      v16bf b2 = b_frag(gT0, nt, kb * 32, RC, lane);
      v16bf b3 = b_frag(gT1, nt, kb * 32, RC, lane);
      fa[0] = wmma_bf(Ap[0][kb], b0, fa[0]);
      fa[1] = wmma_bf(Ap[1][kb], b0, fa[1]);
      fa[0] = wmma_bf(Ac[0][kb], b1, fa[0]);
      fa[1] = wmma_bf(Ac[1][kb], b1, fa[1]);
      ga[0] = wmma_bf(Ap[0][kb], b2, ga[0]);
      ga[1] = wmma_bf(Ap[1][kb], b2, ga[1]);
      ga[0] = wmma_bf(Ac[0][kb], b3, ga[0]);
      ga[1] = wmma_bf(Ac[1][kb], b3, ga[1]);
    }
#pragma unroll
    for (int s = 0; s < 2; ++s)
#pragma unroll
      for (int r = 0; r < 8; ++r) {                   // p = tanh(f)*sigmoid(g)
        const float fv = tanhf(fa[s][r]);
        const float gv = 1.f / (1.f + __expf(-ga[s][r]));
        pl[(s * 16 + r + hi * 8) * RC + nt * 16 + m] = f2bf(fv * gv);
      }
  }

  v16bf Pa[2][4];                                     // reload p in A-layout
#pragma unroll
  for (int s = 0; s < 2; ++s)
#pragma unroll
    for (int kb = 0; kb < 4; ++kb)
      Pa[s][kb] = a_frag(pl, s * 16, kb * 32, RC, lane);

  for (int nt = 0; nt < 8; ++nt) {                    // out += p@skip_w + skip_b
    const float sb = sB[nt * 16 + m];
    v16bf bw[4];
#pragma unroll
    for (int kb = 0; kb < 4; ++kb) bw[kb] = b_frag(skT, nt, kb * 32, RC, lane);
    v8f c[2];
#pragma unroll
    for (int s = 0; s < 2; ++s)
#pragma unroll
      for (int r = 0; r < 8; ++r)
        c[s][r] = out_ac[(size_t)(row0 + s * 16 + r + hi * 8) * RC +
                         nt * 16 + m] + sb;
#pragma unroll
    for (int kb = 0; kb < 4; ++kb) {
      c[0] = wmma_bf(Pa[0][kb], bw[kb], c[0]);
      c[1] = wmma_bf(Pa[1][kb], bw[kb], c[1]);
    }
#pragma unroll
    for (int s = 0; s < 2; ++s)
#pragma unroll
      for (int r = 0; r < 8; ++r)
        out_ac[(size_t)(row0 + s * 16 + r + hi * 8) * RC + nt * 16 + m] =
            c[s][r];
  }
  for (int nt = 0; nt < 8; ++nt) {                    // res += p@res_w + res_b
    const float rb2 = rB[nt * 16 + m];
    v16bf bw[4];
#pragma unroll
    for (int kb = 0; kb < 4; ++kb) bw[kb] = b_frag(reT, nt, kb * 32, RC, lane);
    v8f c[2];
#pragma unroll
    for (int s = 0; s < 2; ++s)
#pragma unroll
      for (int r = 0; r < 8; ++r)
        c[s][r] = res_f[(size_t)(row0 + s * 16 + r + hi * 8) * RC +
                        nt * 16 + m] + rb2;
#pragma unroll
    for (int kb = 0; kb < 4; ++kb) {
      c[0] = wmma_bf(Pa[0][kb], bw[kb], c[0]);
      c[1] = wmma_bf(Pa[1][kb], bw[kb], c[1]);
    }
#pragma unroll
    for (int s = 0; s < 2; ++s)
#pragma unroll
      for (int r = 0; r < 8; ++r) {
        const size_t idx =
            (size_t)(row0 + s * 16 + r + hi * 8) * RC + nt * 16 + m;
        res_f[idx] = c[s][r];
        rout[idx] = f2bf(c[s][r]);                    // next layer's bf16 input
      }
  }
}

// ---------------- post stack: relu -> skip0 -> 31x(32x32) -> final ----------------
__global__ __launch_bounds__(128) void kPost(const float* __restrict__ out_ac,
    const __bf16* __restrict__ s0T, const float* __restrict__ s0B,
    const __bf16* __restrict__ srT, const float* __restrict__ srB,
    const __bf16* __restrict__ fnT, const float* __restrict__ fnB,
    float* __restrict__ Y) {
  __shared__ __bf16 lds[WAVES * MROWS * RC];
  const int lane = threadIdx.x & 31, wave = threadIdx.x >> 5;
  const int m = lane & 15, hi = lane >> 4;
  __bf16* pl = lds + wave * MROWS * RC;
  const int row0 = (blockIdx.x * WAVES + wave) * MROWS;

  for (int i = lane; i < MROWS * RC; i += 32) {
    const float v = out_ac[(size_t)row0 * RC + i];
    pl[i] = f2bf(v > 0.f ? v : 0.f);
  }
  v16bf A[2][4];
#pragma unroll
  for (int s = 0; s < 2; ++s)
#pragma unroll
    for (int kb = 0; kb < 4; ++kb)
      A[s][kb] = a_frag(pl, s * 16, kb * 32, RC, lane);

  v8f v0[2], v1[2];
  {
    const float b0 = s0B[m], b1 = s0B[16 + m];
    v16bf bw0[4], bw1[4];
#pragma unroll
    for (int kb = 0; kb < 4; ++kb) {
      bw0[kb] = b_frag(s0T, 0, kb * 32, RC, lane);
      bw1[kb] = b_frag(s0T, 1, kb * 32, RC, lane);
    }
#pragma unroll
    for (int s = 0; s < 2; ++s)
#pragma unroll
      for (int r = 0; r < 8; ++r) { v0[s][r] = b0; v1[s][r] = b1; }
#pragma unroll
    for (int kb = 0; kb < 4; ++kb) {
      v0[0] = wmma_bf(A[0][kb], bw0[kb], v0[0]);
      v0[1] = wmma_bf(A[1][kb], bw0[kb], v0[1]);
      v1[0] = wmma_bf(A[0][kb], bw1[kb], v1[0]);
      v1[1] = wmma_bf(A[1][kb], bw1[kb], v1[1]);
    }
  }
  for (int i = 0; i < SC - 1; ++i) {
    const __bf16* w = srT + (size_t)i * SC * SC;
    v16bf bw0 = b_frag(w, 0, 0, SC, lane);   // issue weight loads early,
    v16bf bw1 = b_frag(w, 1, 0, SC, lane);   // overlap with LDS round-trip
#pragma unroll
    for (int s = 0; s < 2; ++s)
#pragma unroll
      for (int r = 0; r < 8; ++r) {
        const int rr = s * 16 + r + hi * 8;
        pl[rr * SC + m]      = f2bf(v0[s][r] > 0.f ? v0[s][r] : 0.f);
        pl[rr * SC + 16 + m] = f2bf(v1[s][r] > 0.f ? v1[s][r] : 0.f);
      }
    v16bf A0 = a_frag(pl, 0,  0, SC, lane);
    v16bf A1 = a_frag(pl, 16, 0, SC, lane);
    const float b0 = srB[i * SC + m], b1 = srB[i * SC + 16 + m];
    v8f n0[2], n1[2];
#pragma unroll
    for (int s = 0; s < 2; ++s)
#pragma unroll
      for (int r = 0; r < 8; ++r) { n0[s][r] = b0; n1[s][r] = b1; }
    n0[0] = wmma_bf(A0, bw0, n0[0]);
    n0[1] = wmma_bf(A1, bw0, n0[1]);
    n1[0] = wmma_bf(A0, bw1, n1[0]);
    n1[1] = wmma_bf(A1, bw1, n1[1]);
#pragma unroll
    for (int s = 0; s < 2; ++s) { v0[s] = n0[s]; v1[s] = n1[s]; }
  }
#pragma unroll
  for (int s = 0; s < 2; ++s)
#pragma unroll
    for (int r = 0; r < 8; ++r) {
      const int rr = s * 16 + r + hi * 8;
      pl[rr * SC + m]      = f2bf(v0[s][r] > 0.f ? v0[s][r] : 0.f);
      pl[rr * SC + 16 + m] = f2bf(v1[s][r] > 0.f ? v1[s][r] : 0.f);
    }
  v16bf Af0 = a_frag(pl, 0,  0, SC, lane);
  v16bf Af1 = a_frag(pl, 16, 0, SC, lane);
  // rotated B stream over the 16 output tiles of the final 32->256 matmul
  v16bf bw = b_frag(fnT, 0, 0, SC, lane);
  for (int nt = 0; nt < 16; ++nt) {
    v16bf bn;
    if (nt < 15) bn = b_frag(fnT, nt + 1, 0, SC, lane);
    const float fb = fnB[nt * 16 + m];
    v8f c[2];
#pragma unroll
    for (int s = 0; s < 2; ++s)
#pragma unroll
      for (int r = 0; r < 8; ++r) c[s][r] = fb;
    c[0] = wmma_bf(Af0, bw, c[0]);
    c[1] = wmma_bf(Af1, bw, c[1]);
#pragma unroll
    for (int s = 0; s < 2; ++s)
#pragma unroll
      for (int r = 0; r < 8; ++r)
        Y[(size_t)(row0 + s * 16 + r + hi * 8) * FC + nt * 16 + m] = c[s][r];
    if (nt < 15) bw = bn;
  }
}

// ---------------- host launch ----------------
extern "C" void kernel_launch(void* const* d_in, const int* in_sizes, int n_in,
                              void* d_out, int out_size, void* d_ws, size_t ws_size,
                              hipStream_t stream) {
  const float* X       = (const float*)d_in[0];
  const float* init_w  = (const float*)d_in[1];
  const float* init_b  = (const float*)d_in[2];
  const float* f_w     = (const float*)d_in[3];
  const float* f_b     = (const float*)d_in[4];
  const float* g_w     = (const float*)d_in[5];
  const float* g_b     = (const float*)d_in[6];
  const float* skip_w  = (const float*)d_in[7];
  const float* skip_b  = (const float*)d_in[8];
  const float* res_w   = (const float*)d_in[9];
  const float* res_b   = (const float*)d_in[10];
  const float* skip0_w = (const float*)d_in[11];
  const float* skip0_b = (const float*)d_in[12];
  const float* skr_w   = (const float*)d_in[13];
  const float* skr_b   = (const float*)d_in[14];
  const float* final_w = (const float*)d_in[15];
  const float* final_b = (const float*)d_in[16];

  char* base = (char*)d_ws; size_t off = 0;
  auto carve = [&](size_t bytes) -> char* {
    char* p = base + off; off += (bytes + 255) & ~(size_t)255; return p;
  };
  float*  res_f  = (float*) carve((size_t)BT * RC * 4);
  __bf16* res_a  = (__bf16*)carve((size_t)BT * RC * 2);
  __bf16* res_bb = (__bf16*)carve((size_t)BT * RC * 2);
  float*  out_ac = (float*) carve((size_t)BT * RC * 4);
  __bf16* Xbf    = (__bf16*)carve((size_t)BT * FIN * 2);
  __bf16* initT  = (__bf16*)carve((size_t)32 * RC * FIN * 2);
  __bf16* fT     = (__bf16*)carve((size_t)NLAYER * 2 * RC * RC * 2);
  __bf16* gT     = (__bf16*)carve((size_t)NLAYER * 2 * RC * RC * 2);
  __bf16* skT    = (__bf16*)carve((size_t)NLAYER * RC * RC * 2);
  __bf16* reT    = (__bf16*)carve((size_t)NLAYER * RC * RC * 2);
  __bf16* s0T    = (__bf16*)carve((size_t)SC * RC * 2);
  __bf16* srT    = (__bf16*)carve((size_t)(SC - 1) * SC * SC * 2);
  __bf16* fnT    = (__bf16*)carve((size_t)FC * SC * 2);

  kCvt<<<2048, 256, 0, stream>>>(X, Xbf, BT * FIN);
  kTr<<<dim3(32, 32), 256, 0, stream>>>(init_w, initT, FIN, RC);
  kTr<<<dim3(32, 40), 256, 0, stream>>>(f_w, fT, RC, RC);
  kTr<<<dim3(32, 40), 256, 0, stream>>>(g_w, gT, RC, RC);
  kTr<<<dim3(32, 20), 256, 0, stream>>>(skip_w, skT, RC, RC);
  kTr<<<dim3(32, 20), 256, 0, stream>>>(res_w, reT, RC, RC);
  kTr<<<dim3(4, 1),  256, 0, stream>>>(skip0_w, s0T, RC, SC);
  kTr<<<dim3(2, 31), 256, 0, stream>>>(skr_w, srT, SC, SC);
  kTr<<<dim3(8, 1),  256, 0, stream>>>(final_w, fnT, SC, FC);

  const int NBLK = BT / (MROWS * WAVES);   // 512
  kInit<<<NBLK, 32 * WAVES, 0, stream>>>(Xbf, initT, init_b, res_f, res_a, out_ac);

  static const int DIL[NLAYER] = {1, 2, 4, 8, 16, 32, 64, 128, 256, 512,
                                  1, 2, 4, 8, 16, 32, 64, 128, 256, 512};
  const __bf16* rin = res_a; __bf16* rout = res_bb;
  for (int i = 0; i < NLAYER; ++i) {
    kLayer<<<NBLK, 32 * WAVES, 0, stream>>>(
        rin, res_f, rout, out_ac,
        fT + (size_t)(2 * i) * RC * RC, fT + (size_t)(2 * i + 1) * RC * RC,
        gT + (size_t)(2 * i) * RC * RC, gT + (size_t)(2 * i + 1) * RC * RC,
        skT + (size_t)i * RC * RC, reT + (size_t)i * RC * RC,
        f_b + i * RC, g_b + i * RC, skip_b + i * RC, res_b + i * RC, DIL[i]);
    const __bf16* t = rin; rin = rout; rout = (__bf16*)t;   // ping-pong
  }

  kPost<<<NBLK, 32 * WAVES, 0, stream>>>(out_ac, s0T, skip0_b, srT, skr_b,
                                         fnT, final_b, (float*)d_out);
}
